// Qwen3MoeSparseMoeBlockParallel_74345883894649
// MI455X (gfx1250) — compile-verified, matrix operand order checked by reading, not run on hardware
//
#include <hip/hip_runtime.h>
#include <hip/hip_bf16.h>
#include <math.h>

// ---- problem constants (match reference) ----
#define E_    16
#define TOPK  4
#define H_    2048
#define I_    768
#define R_    16
#define T_    2048          // B*S = 2*1024
#define ALPHA_ 2.0f         // 32 / R
#define MT    128           // M-tile rows (tokens) for the grouped GEMMs

typedef __bf16 bf16_t;
typedef __attribute__((ext_vector_type(16))) __bf16 v16bf;
typedef __attribute__((ext_vector_type(8)))  __bf16 v8bf;
typedef __attribute__((ext_vector_type(4)))  __bf16 v4bf;
typedef __attribute__((ext_vector_type(8)))  float  v8f;

#define LDSTR 40   // padded K-stride (halfs); 80B rows keep 16B alignment
#define OSTR  72   // padded col-stride (halfs) for the output staging tile

static __device__ __forceinline__ v16bf cat8(v8bf lo, v8bf hi) {
  return __builtin_shufflevector(lo, hi, 0,1,2,3,4,5,6,7,8,9,10,11,12,13,14,15);
}

// 16x32 (rows x K) bf16 fragment from LDS, ISA A/B-operand layout:
//   lanes 0-15 : row = lane,    K {0..7, 16..23}
//   lanes 16-31: row = lane-16, K {8..15, 24..31}
static __device__ __forceinline__ v16bf frag_lds(const bf16_t* base, int sub, int lane) {
  const bf16_t* p = base + (size_t)(sub + (lane & 15)) * LDSTR + ((lane < 16) ? 0 : 8);
  return cat8(*(const v8bf*)p, *(const v8bf*)(p + 16));
}

static __device__ __forceinline__ v8f wmma_bf16(v16bf a, v16bf b, v8f c) {
  return __builtin_amdgcn_wmma_f32_16x16x32_bf16(false, a, false, b, (short)0, c, false, false);
}

static __device__ __forceinline__ void store4(bf16_t* dst, float a, float b, float c, float d) {
  v4bf v; v[0] = (bf16_t)a; v[1] = (bf16_t)b; v[2] = (bf16_t)c; v[3] = (bf16_t)d;
  *(v4bf*)dst = v;       // ds_store_b64, conversions fuse into v_cvt_pk_bf16_f32
}
static __device__ __forceinline__ void zero4(bf16_t* dst) {
  v4bf z = {}; *(v4bf*)dst = z;
}

// ---------------------------------------------------------------------------
// 0) zero output + expert counters
__global__ void k_zero(float* __restrict__ out, int n, int* __restrict__ counts) {
  int i = blockIdx.x * 256 + threadIdx.x;
  if (i < n) out[i] = 0.0f;
  if (i < E_) counts[i] = 0;
}

// 1) fp32 -> bf16 activation staging
__global__ void k_cvt(const float* __restrict__ x, bf16_t* __restrict__ y, int n) {
  int i = blockIdx.x * 256 + threadIdx.x;
  if (i < n) y[i] = (bf16_t)x[i];
}

// ---------------------------------------------------------------------------
// 2) router: logits, fp32 softmax, top-4, renorm, build per-expert token lists.
__global__ void k_router(const float* __restrict__ x, const float* __restrict__ rw,
                         float* __restrict__ logits, int* __restrict__ counts,
                         int* __restrict__ ids, float* __restrict__ wts) {
  int lane = threadIdx.x & 31;
  int wid  = threadIdx.x >> 5;
  int t = blockIdx.x * 8 + wid;

  const float* xr = x + (size_t)t * H_;
  float xreg[H_ / 32];
  #pragma unroll
  for (int j = 0; j < H_ / 32; ++j) xreg[j] = xr[lane + 32 * j];

  float lg[E_];
  #pragma unroll
  for (int e = 0; e < E_; ++e) {
    const float* we = rw + (size_t)e * H_;
    float acc = 0.0f;
    #pragma unroll 8
    for (int j = 0; j < H_ / 32; ++j) acc += xreg[j] * we[lane + 32 * j];
    #pragma unroll
    for (int m = 16; m >= 1; m >>= 1) acc += __shfl_xor(acc, m, 32);
    lg[e] = acc;
  }

  if (lane < E_) logits[(size_t)t * E_ + lane] = lg[lane];   // raw logits output

  float mx = lg[0];
  #pragma unroll
  for (int e = 1; e < E_; ++e) mx = fmaxf(mx, lg[e]);
  float p[E_]; float sum = 0.0f;
  #pragma unroll
  for (int e = 0; e < E_; ++e) { p[e] = __expf(lg[e] - mx); sum += p[e]; }
  float inv = 1.0f / sum;
  #pragma unroll
  for (int e = 0; e < E_; ++e) p[e] *= inv;

  int sel[TOPK]; float wk[TOPK]; float wsum = 0.0f;
  #pragma unroll
  for (int k = 0; k < TOPK; ++k) {
    int bj = 0; float bv = p[0];
    #pragma unroll
    for (int e = 1; e < E_; ++e) { if (p[e] > bv) { bv = p[e]; bj = e; } }
    sel[k] = bj; wk[k] = bv; p[bj] = -1.0f; wsum += bv;
  }
  float rinv = 1.0f / wsum;

  if (lane < TOPK) {
    int eSel = sel[lane];
    int pos = atomicAdd(&counts[eSel], 1);
    ids[(size_t)eSel * T_ + pos] = t;
    wts[(size_t)eSel * T_ + pos] = wk[lane] * rinv;
  }
}

// ---------------------------------------------------------------------------
// 3) LoRA-A prepass: xga[e,t,:] = ALPHA * (x[t,:] @ gate_A[e]); same for up_A.
__global__ __launch_bounds__(128)
void k_lora_a(const bf16_t* __restrict__ xb,
              const float* __restrict__ gA, const float* __restrict__ uA,
              bf16_t* __restrict__ xga, bf16_t* __restrict__ xua) {
  __shared__ bf16_t lBg[16 * LDSTR];
  __shared__ bf16_t lBu[16 * LDSTR];
  int tid = threadIdx.x; int lane = tid & 31; int w = tid >> 5;
  int t0 = blockIdx.x * 64; int e = blockIdx.y;
  const float* gAe = gA + (size_t)e * H_ * R_;
  const float* uAe = uA + (size_t)e * H_ * R_;
  v8f accG = {}; v8f accU = {};

  int fn = tid & 15, fk = tid >> 4;             // 16n x 8 k-groups, 1 item/thread
  for (int h0 = 0; h0 < H_; h0 += 32) {
    const float* pg = gAe + (size_t)(h0 + fk * 4) * R_ + fn;
    const float* pu = uAe + (size_t)(h0 + fk * 4) * R_ + fn;
    store4(&lBg[fn * LDSTR + fk * 4], pg[0], pg[R_], pg[2 * R_], pg[3 * R_]);
    store4(&lBu[fn * LDSTR + fk * 4], pu[0], pu[R_], pu[2 * R_], pu[3 * R_]);
    __syncthreads();
    const bf16_t* pa = xb + (size_t)(t0 + w * 16 + (lane & 15)) * H_ + h0 + ((lane < 16) ? 0 : 8);
    v16bf a = cat8(*(const v8bf*)pa, *(const v8bf*)(pa + 16));
    accG = wmma_bf16(a, frag_lds(lBg, 0, lane), accG);
    accU = wmma_bf16(a, frag_lds(lBu, 0, lane), accU);
    __syncthreads();
  }

  int mBase = w * 16 + ((lane < 16) ? 0 : 8);
  int n = lane & 15;
  #pragma unroll
  for (int r = 0; r < 8; ++r) {
    int row = t0 + mBase + r;
    xga[((size_t)e * T_ + row) * R_ + n] = (bf16_t)(ALPHA_ * accG[r]);
    xua[((size_t)e * T_ + row) * R_ + n] = (bf16_t)(ALPHA_ * accU[r]);
  }
}

// ---------------------------------------------------------------------------
// 4) gate/up grouped GEMM (+ LoRA-B, + SiLU*up) -> inter (bf16)
//    Grid: (T/128, I/64, E). 256 threads = 8 waves, wave w owns rows 16w..16w+15.
//    Double-buffered LDS: one barrier per K-step; tile i+1 fills while tile i
//    feeds the WMMAs.
__global__ __launch_bounds__(256)
void k_gate_up(const bf16_t* __restrict__ xb,
               const float* __restrict__ gP, const float* __restrict__ uP,
               const float* __restrict__ gB, const float* __restrict__ uB,
               const bf16_t* __restrict__ xga, const bf16_t* __restrict__ xua,
               const int* __restrict__ counts, const int* __restrict__ ids,
               bf16_t* __restrict__ inter) {
  int e = blockIdx.z;
  int cnt = counts[e];
  int mt0 = blockIdx.x * MT;
  if (mt0 >= cnt) return;
  int i0 = blockIdx.y * 64;

  __shared__ bf16_t lBg[2][64 * LDSTR];
  __shared__ bf16_t lBu[2][64 * LDSTR];
  __shared__ bf16_t lOut[MT * OSTR];
  __shared__ int tokLds[MT];

  int tid = threadIdx.x; int lane = tid & 31; int w = tid >> 5;
  if (tid < MT) {
    int slot = mt0 + tid;
    tokLds[tid] = ids[(size_t)e * T_ + ((slot < cnt) ? slot : mt0)];
  }
  __syncthreads();
  int myTok = tokLds[w * 16 + (lane & 15)];

  const float* gPe = gP + (size_t)e * H_ * I_;
  const float* uPe = uP + (size_t)e * H_ * I_;
  v8f accG[4] = {{}, {}, {}, {}};
  v8f accU[4] = {{}, {}, {}, {}};

  // fill assignment: 64n x 8 k-groups = 512 items, 2 per thread
  float rg[2][4], ru[2][4];
  auto loadB = [&](int h0) {
    #pragma unroll
    for (int j = 0; j < 2; ++j) {
      int idx = j * 256 + tid; int n = idx & 63; int kg = idx >> 6;
      const float* pg = gPe + (size_t)(h0 + kg * 4) * I_ + i0 + n;
      const float* pu = uPe + (size_t)(h0 + kg * 4) * I_ + i0 + n;
      #pragma unroll
      for (int q = 0; q < 4; ++q) { rg[j][q] = pg[(size_t)q * I_]; ru[j][q] = pu[(size_t)q * I_]; }
      if (h0 + 32 < H_) {                     // global_prefetch_b8 one tile further
        __builtin_prefetch(pg + (size_t)32 * I_, 0, 0);
        __builtin_prefetch(pu + (size_t)32 * I_, 0, 0);
      }
    }
  };
  auto storeB = [&](int b) {
    #pragma unroll
    for (int j = 0; j < 2; ++j) {
      int idx = j * 256 + tid; int n = idx & 63; int kg = idx >> 6;
      store4(&lBg[b][n * LDSTR + kg * 4], rg[j][0], rg[j][1], rg[j][2], rg[j][3]);
      store4(&lBu[b][n * LDSTR + kg * 4], ru[j][0], ru[j][1], ru[j][2], ru[j][3]);
    }
  };

  // prologue: buf0 = tile0, regs = tile1
  loadB(0);
  storeB(0);
  loadB(32);
  __syncthreads();
  for (int h0 = 0; h0 < H_; h0 += 32) {
    int b = (h0 >> 5) & 1;
    if (h0 + 32 < H_) storeB(b ^ 1);        // fill other buffer while computing
    const bf16_t* pa = xb + (size_t)myTok * H_ + h0 + ((lane < 16) ? 0 : 8);
    v16bf a = cat8(*(const v8bf*)pa, *(const v8bf*)(pa + 16));
    #pragma unroll
    for (int ns = 0; ns < 4; ++ns) {
      accG[ns] = wmma_bf16(a, frag_lds(lBg[b], ns * 16, lane), accG[ns]);
      accU[ns] = wmma_bf16(a, frag_lds(lBu[b], ns * 16, lane), accU[ns]);
    }
    if (h0 + 64 < H_) loadB(h0 + 64);       // global loads for tile i+2
    __syncthreads();                        // one barrier per K-step
  }

  // fused LoRA-B (gate+up): K padded 16->32 with zeros; single barrier.
  {
    const float* gBe = gB + (size_t)e * R_ * I_;
    const float* uBe = uB + (size_t)e * R_ * I_;
    int n = tid & 63; int kg = tid >> 6;    // kg 0..3 => k 0..15
    const float* pg = gBe + (size_t)(kg * 4) * I_ + i0 + n;
    const float* pu = uBe + (size_t)(kg * 4) * I_ + i0 + n;
    store4(&lBg[0][n * LDSTR + kg * 4], pg[0], pg[(size_t)I_], pg[2 * (size_t)I_], pg[3 * (size_t)I_]);
    store4(&lBu[0][n * LDSTR + kg * 4], pu[0], pu[(size_t)I_], pu[2 * (size_t)I_], pu[3 * (size_t)I_]);
    zero4(&lBg[0][n * LDSTR + 16 + kg * 4]);
    zero4(&lBu[0][n * LDSTR + 16 + kg * 4]);
    __syncthreads();
    v8bf z = {};
    const bf16_t* pag = xga + ((size_t)e * T_ + myTok) * R_ + ((lane < 16) ? 0 : 8);
    const bf16_t* pau = xua + ((size_t)e * T_ + myTok) * R_ + ((lane < 16) ? 0 : 8);
    v16bf ag = cat8(*(const v8bf*)pag, z);
    v16bf au = cat8(*(const v8bf*)pau, z);
    #pragma unroll
    for (int ns = 0; ns < 4; ++ns) {
      accG[ns] = wmma_bf16(ag, frag_lds(lBg[0], ns * 16, lane), accG[ns]);
      accU[ns] = wmma_bf16(au, frag_lds(lBu[0], ns * 16, lane), accU[ns]);
    }
  }

  // SiLU(g)*u -> LDS staging, then coalesced b128 stores to inter.
  int mBase = w * 16 + ((lane < 16) ? 0 : 8);
  #pragma unroll
  for (int ns = 0; ns < 4; ++ns) {
    #pragma unroll
    for (int r = 0; r < 8; ++r) {
      float g = accG[ns][r], u = accU[ns][r];
      float val = (g / (1.0f + __expf(-g))) * u;
      lOut[(size_t)(mBase + r) * OSTR + ns * 16 + (lane & 15)] = (bf16_t)val;
    }
  }
  __syncthreads();
  // 128 rows x 64 halfs: thread t -> row t>>1, col base (t&1)*32, 4 x v8bf
  {
    int row = tid >> 1; int cb = (tid & 1) * 32;
    bf16_t* dst = inter + ((size_t)e * T_ + mt0 + row) * I_ + i0 + cb;
    const bf16_t* src = &lOut[(size_t)row * OSTR + cb];
    #pragma unroll
    for (int j = 0; j < 4; ++j)
      *(v8bf*)(dst + 8 * j) = *(const v8bf*)(src + 8 * j);
  }
}

// ---------------------------------------------------------------------------
// 5) down grouped GEMM (+ down LoRA) + weighted combine via fp32 atomics.
//    Grid: (T/128, H/64, E). 256 threads = 8 waves. Double-buffered LDS.
__global__ __launch_bounds__(256)
void k_down(const bf16_t* __restrict__ inter,
            const float* __restrict__ dP, const float* __restrict__ dA,
            const float* __restrict__ dB,
            const int* __restrict__ counts, const int* __restrict__ ids,
            const float* __restrict__ wts, float* __restrict__ out) {
  int e = blockIdx.z;
  int cnt = counts[e];
  int mt0 = blockIdx.x * MT;
  if (mt0 >= cnt) return;
  int h0c = blockIdx.y * 64;

  __shared__ bf16_t lBd[2][64 * LDSTR];
  __shared__ bf16_t lBa[2][16 * LDSTR];
  __shared__ bf16_t lAa[MT * LDSTR];
  __shared__ int   tokLds[MT];
  __shared__ float wtLds[MT];

  int tid = threadIdx.x; int lane = tid & 31; int w = tid >> 5;
  if (tid < MT) {
    int slot = mt0 + tid;
    int ok = slot < cnt;
    tokLds[tid] = ids[(size_t)e * T_ + (ok ? slot : mt0)];
    wtLds[tid]  = ok ? wts[(size_t)e * T_ + slot] : 0.0f;
  }
  __syncthreads();

  const float* dPe = dP + (size_t)e * I_ * H_;
  const float* dAe = dA + (size_t)e * I_ * R_;
  v8f accD[4] = {{}, {}, {}, {}};
  v8f accA = {};

  float rd[2][4], ra[4];
  int an = tid & 15, ak = tid >> 4;          // lBa: 16n x 8kg, threads 0..127
  auto loadB = [&](int i0) {
    #pragma unroll
    for (int j = 0; j < 2; ++j) {
      int idx = j * 256 + tid; int n = idx & 63; int kg = idx >> 6;
      const float* p = dPe + (size_t)(i0 + kg * 4) * H_ + h0c + n;
      #pragma unroll
      for (int q = 0; q < 4; ++q) rd[j][q] = p[(size_t)q * H_];
      if (i0 + 32 < I_) __builtin_prefetch(p + (size_t)32 * H_, 0, 0);
    }
    if (tid < 128) {
      const float* p = dAe + (size_t)(i0 + ak * 4) * R_ + an;
      #pragma unroll
      for (int q = 0; q < 4; ++q) ra[q] = p[(size_t)q * R_];
    }
  };
  auto storeB = [&](int b) {
    #pragma unroll
    for (int j = 0; j < 2; ++j) {
      int idx = j * 256 + tid; int n = idx & 63; int kg = idx >> 6;
      store4(&lBd[b][n * LDSTR + kg * 4], rd[j][0], rd[j][1], rd[j][2], rd[j][3]);
    }
    if (tid < 128) store4(&lBa[b][an * LDSTR + ak * 4], ra[0], ra[1], ra[2], ra[3]);
  };

  loadB(0);
  storeB(0);
  loadB(32);
  __syncthreads();
  for (int i0 = 0; i0 < I_; i0 += 32) {
    int b = (i0 >> 5) & 1;
    if (i0 + 32 < I_) storeB(b ^ 1);
    const bf16_t* pa = inter + ((size_t)e * T_ + mt0 + w * 16 + (lane & 15)) * I_
                             + i0 + ((lane < 16) ? 0 : 8);
    v16bf a = cat8(*(const v8bf*)pa, *(const v8bf*)(pa + 16));
    #pragma unroll
    for (int ns = 0; ns < 4; ++ns)
      accD[ns] = wmma_bf16(a, frag_lds(lBd[b], ns * 16, lane), accD[ns]);
    accA = wmma_bf16(a, frag_lds(lBa[b], 0, lane), accA);
    if (i0 + 64 < I_) loadB(i0 + 64);
    __syncthreads();
  }

  // down LoRA: (ALPHA * accA) @ down_B via LDS round-trip of accA as an A-tile.
  {
    int mBase = w * 16 + ((lane < 16) ? 0 : 8);
    #pragma unroll
    for (int r = 0; r < 8; ++r)
      lAa[(size_t)(mBase + r) * LDSTR + (lane & 15)] = (bf16_t)(ALPHA_ * accA[r]);
    #pragma unroll
    for (int j = 0; j < 2; ++j) {            // zero K 16..31 over 128 rows
      int idx = j * 256 + tid; int row = idx >> 2; int pk = idx & 3;
      zero4(&lAa[(size_t)row * LDSTR + 16 + pk * 4]);
    }
    const float* dBe = dB + (size_t)e * R_ * H_;
    int n = tid & 63; int kg = tid >> 6;     // kg 0..3 => k 0..15
    const float* p = dBe + (size_t)(kg * 4) * H_ + h0c + n;
    store4(&lBd[0][n * LDSTR + kg * 4], p[0], p[(size_t)H_], p[2 * (size_t)H_], p[3 * (size_t)H_]);
    zero4(&lBd[0][n * LDSTR + 16 + kg * 4]);
    __syncthreads();
    v16bf a = frag_lds(lAa, w * 16, lane);
    #pragma unroll
    for (int ns = 0; ns < 4; ++ns)
      accD[ns] = wmma_bf16(a, frag_lds(lBd[0], ns * 16, lane), accD[ns]);
  }

  // combine: out[tok] += weight * d   (global_atomic_add_f32)
  int validRows = cnt - mt0; if (validRows > MT) validRows = MT;
  int mBase = w * 16 + ((lane < 16) ? 0 : 8);
  #pragma unroll
  for (int r = 0; r < 8; ++r) {
    int m = mBase + r;
    if (m < validRows) {
      int tok = tokLds[m];
      float wt = wtLds[m];
      #pragma unroll
      for (int ns = 0; ns < 4; ++ns) {
        unsafeAtomicAdd(&out[(size_t)tok * H_ + h0c + ns * 16 + (lane & 15)],
                        wt * accD[ns][r]);
      }
    }
  }
}

// ---------------------------------------------------------------------------
extern "C" void kernel_launch(void* const* d_in, const int* in_sizes, int n_in,
                              void* d_out, int out_size, void* d_ws, size_t ws_size,
                              hipStream_t stream) {
  const float* x  = (const float*)d_in[0];
  const float* rw = (const float*)d_in[1];
  const float* gP = (const float*)d_in[2];
  const float* uP = (const float*)d_in[3];
  const float* dP = (const float*)d_in[4];
  const float* gA = (const float*)d_in[5];
  const float* gB = (const float*)d_in[6];
  const float* uA = (const float*)d_in[7];
  const float* uB = (const float*)d_in[8];
  const float* dA = (const float*)d_in[9];
  const float* dB = (const float*)d_in[10];
  float* out = (float*)d_out;

  // workspace carve-out (~60 MB): counts | ids | wts | x_bf16 | xga | xua | inter
  char* ws = (char*)d_ws;
  size_t off = 0;
  auto carve = [&](size_t bytes) -> void* {
    void* p = ws + off;
    off = (off + bytes + 255) & ~(size_t)255;
    return p;
  };
  int*    counts = (int*)   carve(E_ * sizeof(int));
  int*    ids    = (int*)   carve((size_t)E_ * T_ * sizeof(int));
  float*  wts    = (float*) carve((size_t)E_ * T_ * sizeof(float));
  bf16_t* xb     = (bf16_t*)carve((size_t)T_ * H_ * sizeof(bf16_t));
  bf16_t* xga    = (bf16_t*)carve((size_t)E_ * T_ * R_ * sizeof(bf16_t));
  bf16_t* xua    = (bf16_t*)carve((size_t)E_ * T_ * R_ * sizeof(bf16_t));
  bf16_t* inter  = (bf16_t*)carve((size_t)E_ * T_ * I_ * sizeof(bf16_t));
  (void)ws_size; (void)in_sizes; (void)n_in; (void)out_size;

  const int nTH = T_ * H_;
  k_zero  <<<(nTH + 255) / 256, 256, 0, stream>>>(out, nTH, counts);
  k_cvt   <<<(nTH + 255) / 256, 256, 0, stream>>>(x, xb, nTH);
  k_router<<<T_ / 8, 256, 0, stream>>>(x, rw, out + (size_t)nTH, counts, ids, wts);
  k_lora_a<<<dim3(T_ / 64, E_), 128, 0, stream>>>(xb, gA, uA, xga, xua);
  k_gate_up<<<dim3(T_ / MT, I_ / 64, E_), 256, 0, stream>>>(xb, gP, uP, gB, uB, xga, xua,
                                                            counts, ids, inter);
  k_down  <<<dim3(T_ / MT, H_ / 64, E_), 256, 0, stream>>>(inter, dP, dA, dB,
                                                           counts, ids, wts, out);
}
